// SwinTransformer_59889023975571
// MI455X (gfx1250) — compile-verified, matrix operand order checked by reading, not run on hardware
//
#include <hip/hip_runtime.h>
#include <hip/hip_bf16.h>
#include <math.h>

// ---------------------------------------------------------------------------
// Swin Transformer forward for MI455X (gfx1250, wave32, WMMA).
// All GEMMs + attention use v_wmma_f32_16x16x32_f16 (f16 in, f32 accum).
// GEMM: 32x32 tile per wave (4 WMMAs / k-step), branch-free K loop via
// row-index clamping (out-of-range rows never stored, so clamping is safe).
// ---------------------------------------------------------------------------

typedef __attribute__((ext_vector_type(16))) _Float16 v16h;
typedef __attribute__((ext_vector_type(8)))  _Float16 v8h;
typedef __attribute__((ext_vector_type(8)))  float    v8f;
typedef __attribute__((ext_vector_type(4)))  float    v4f;

#define IMG  640
#define HP   160          // tokens per side after 4x4 patches
#define EMB  192

// ----- fragment builders ----------------------------------------------------
// 16-bit A/B fragment layout (ISA 7.12.2): lane&15 = row (A:M, B:N);
// halves e<8  -> K = e   + 8*grp ; halves e>=8 -> K = e+8 + 8*grp
// => two contiguous 8-half chunks at offsets 8*grp and 16+8*grp of the row.

__device__ inline v16h frag_from_f32(const float* __restrict__ rowp, int k0, int grp) {
  const float* p = rowp + k0 + (grp << 3);
  v4f a = *(const v4f*)(p);
  v4f b = *(const v4f*)(p + 4);
  v4f c = *(const v4f*)(p + 16);
  v4f d = *(const v4f*)(p + 20);
  v16h f;
  f[0]=(_Float16)a[0]; f[1]=(_Float16)a[1]; f[2]=(_Float16)a[2]; f[3]=(_Float16)a[3];
  f[4]=(_Float16)b[0]; f[5]=(_Float16)b[1]; f[6]=(_Float16)b[2]; f[7]=(_Float16)b[3];
  f[8]=(_Float16)c[0]; f[9]=(_Float16)c[1]; f[10]=(_Float16)c[2]; f[11]=(_Float16)c[3];
  f[12]=(_Float16)d[0]; f[13]=(_Float16)d[1]; f[14]=(_Float16)d[2]; f[15]=(_Float16)d[3];
  return f;
}

__device__ inline v16h frag_from_h(const _Float16* __restrict__ rowp, int grp) {
  v8h a = *(const v8h*)(rowp + (grp << 3));
  v8h b = *(const v8h*)(rowp + 16 + (grp << 3));
  v16h f;
#pragma unroll
  for (int i = 0; i < 8; ++i) { f[i] = a[i]; f[i + 8] = b[i]; }
  return f;
}

__device__ inline int iclamp(int v, int hi) { return v < hi ? v : hi; }

// ----- generic GEMM:  C[M,N] = A[M,K] @ W[N,K]^T (+bias)(+GELU)(+res) -------
// One wave computes a 32(M) x 32(N) tile: 4 accumulators, 4 WMMAs per k-step.
__global__ __launch_bounds__(128)
void gemm_wmma_kernel(const float* __restrict__ A, const float* __restrict__ W,
                      const float* __restrict__ bias, const float* __restrict__ res,
                      float* __restrict__ Co, int M, int Nn, int K, int fuse_gelu) {
  const int lane = threadIdx.x;
  const int mrow = lane & 15, grp = lane >> 4;
  const int tn = (blockIdx.x * 4 + threadIdx.y) * 32;
  const int tm = blockIdx.y * 32;
  if (tn >= Nn) return;

  // clamp out-of-range rows to last valid row: contributions land only in
  // output rows/cols that are never stored, so this is safe and branch-free.
  const float* ap0 = A + (size_t)iclamp(tm + mrow,       M - 1) * K;
  const float* ap1 = A + (size_t)iclamp(tm + 16 + mrow,  M - 1) * K;
  const float* bp0 = W + (size_t)iclamp(tn + mrow,       Nn - 1) * K;
  const float* bp1 = W + (size_t)iclamp(tn + 16 + mrow,  Nn - 1) * K;

  v8f acc00, acc01, acc10, acc11;
#pragma unroll
  for (int i = 0; i < 8; ++i) { acc00[i] = 0.f; acc01[i] = 0.f; acc10[i] = 0.f; acc11[i] = 0.f; }

  for (int k0 = 0; k0 < K; k0 += 32) {
    v16h a0 = frag_from_f32(ap0, k0, grp);
    v16h a1 = frag_from_f32(ap1, k0, grp);
    v16h b0 = frag_from_f32(bp0, k0, grp);
    v16h b1 = frag_from_f32(bp1, k0, grp);
    acc00 = __builtin_amdgcn_wmma_f32_16x16x32_f16(false, a0, false, b0, (short)0, acc00, false, false);
    acc01 = __builtin_amdgcn_wmma_f32_16x16x32_f16(false, a0, false, b1, (short)0, acc01, false, false);
    acc10 = __builtin_amdgcn_wmma_f32_16x16x32_f16(false, a1, false, b0, (short)0, acc10, false, false);
    acc11 = __builtin_amdgcn_wmma_f32_16x16x32_f16(false, a1, false, b1, (short)0, acc11, false, false);
  }

  const int c0 = tn + mrow, c1 = tn + 16 + mrow;
  const float bc0 = (bias && c0 < Nn) ? bias[c0] : 0.f;
  const float bc1 = (bias && c1 < Nn) ? bias[c1] : 0.f;
#pragma unroll
  for (int vg = 0; vg < 8; ++vg) {
    int r0 = tm + vg + 8 * grp;
    int r1 = r0 + 16;
    float v00 = acc00[vg] + bc0, v01 = acc01[vg] + bc1;
    float v10 = acc10[vg] + bc0, v11 = acc11[vg] + bc1;
    if (fuse_gelu) {
      v00 = 0.5f * v00 * (1.f + erff(v00 * 0.70710678118654752f));
      v01 = 0.5f * v01 * (1.f + erff(v01 * 0.70710678118654752f));
      v10 = 0.5f * v10 * (1.f + erff(v10 * 0.70710678118654752f));
      v11 = 0.5f * v11 * (1.f + erff(v11 * 0.70710678118654752f));
    }
    if (r0 < M) {
      if (c0 < Nn) Co[(size_t)r0 * Nn + c0] = v00 + (res ? res[(size_t)r0 * Nn + c0] : 0.f);
      if (c1 < Nn) Co[(size_t)r0 * Nn + c1] = v01 + (res ? res[(size_t)r0 * Nn + c1] : 0.f);
    }
    if (r1 < M) {
      if (c0 < Nn) Co[(size_t)r1 * Nn + c0] = v10 + (res ? res[(size_t)r1 * Nn + c0] : 0.f);
      if (c1 < Nn) Co[(size_t)r1 * Nn + c1] = v11 + (res ? res[(size_t)r1 * Nn + c1] : 0.f);
    }
  }
}

// ----- flash window attention ----------------------------------------------
// One wave per (window, head, 16-query tile). K padded 24->32 with zeros.
__global__ __launch_bounds__(32)
void flash_attn_kernel(const _Float16* __restrict__ qh, const _Float16* __restrict__ kh,
                       const _Float16* __restrict__ vT, const float* __restrict__ rpb,
                       float* __restrict__ out,
                       int N, int ws, int nH, int nWside, int Hg, int shift, int C) {
  __shared__ __align__(16) _Float16 lds_p[16 * 32];
  const int lane = threadIdx.x;
  const int mrow = lane & 15, grp = lane >> 4;
  const int w = blockIdx.z, h = blockIdx.y, qt = blockIdx.x;
  const int wy = w / nWside, wx = w % nWside;
  const int hd = C / nH;              // 24
  const size_t wh = (size_t)w * nH + h;
  const _Float16* qb = qh + wh * N * 32;
  const _Float16* kb = kh + wh * N * 32;
  const _Float16* vb = vT + wh * 32 * N;

  const v16h qf = frag_from_h(qb + (size_t)(qt * 16 + mrow) * 32, grp);

  int qy[8], qx[8], rq[8];
#pragma unroll
  for (int vg = 0; vg < 8; ++vg) {
    int r = qt * 16 + vg + 8 * grp;
    qy[vg] = r / ws; qx[vg] = r % ws;
    if (shift > 0) {
      int gy = wy * ws + qy[vg], gx = wx * ws + qx[vg];
      int ry = (gy < Hg - ws) ? 0 : (gy < Hg - shift ? 1 : 2);
      int rx = (gx < Hg - ws) ? 0 : (gx < Hg - shift ? 1 : 2);
      rq[vg] = 3 * ry + rx;
    } else rq[vg] = 0;
  }

  v8f acc0, acc1, zacc;
  float mrun[8], lrun[8];
#pragma unroll
  for (int vg = 0; vg < 8; ++vg) {
    acc0[vg] = 0.f; acc1[vg] = 0.f; zacc[vg] = 0.f;
    mrun[vg] = -1e30f; lrun[vg] = 0.f;
  }

  const int span = 2 * ws - 1;
  for (int j0 = 0; j0 < N; j0 += 32) {
    const int m0 = j0 + mrow, m1 = m0 + 16;
    v16h kf0 = frag_from_h(kb + (size_t)m0 * 32, grp);
    v16h kf1 = frag_from_h(kb + (size_t)m1 * 32, grp);
    v8f s0 = __builtin_amdgcn_wmma_f32_16x16x32_f16(false, qf, false, kf0, (short)0, zacc, false, false);
    v8f s1 = __builtin_amdgcn_wmma_f32_16x16x32_f16(false, qf, false, kf1, (short)0, zacc, false, false);

    const int ky0 = m0 / ws, kx0 = m0 % ws;
    const int ky1 = m1 / ws, kx1 = m1 % ws;
    int rk0 = 0, rk1 = 0;
    if (shift > 0) {
      int g, a, b;
      g = wy * ws + ky0; a = (g < Hg - ws) ? 0 : (g < Hg - shift ? 1 : 2);
      g = wx * ws + kx0; b = (g < Hg - ws) ? 0 : (g < Hg - shift ? 1 : 2);
      rk0 = 3 * a + b;
      g = wy * ws + ky1; a = (g < Hg - ws) ? 0 : (g < Hg - shift ? 1 : 2);
      g = wx * ws + kx1; b = (g < Hg - ws) ? 0 : (g < Hg - shift ? 1 : 2);
      rk1 = 3 * a + b;
    }

#pragma unroll
    for (int vg = 0; vg < 8; ++vg) {
      float sv0 = s0[vg] + rpb[((qy[vg] - ky0 + ws - 1) * span + (qx[vg] - kx0 + ws - 1)) * nH + h];
      float sv1 = s1[vg] + rpb[((qy[vg] - ky1 + ws - 1) * span + (qx[vg] - kx1 + ws - 1)) * nH + h];
      if (shift > 0) {
        if (rq[vg] != rk0) sv0 -= 100.f;
        if (rq[vg] != rk1) sv1 -= 100.f;
      }
      float t = fmaxf(sv0, sv1);
      t = fmaxf(t, __shfl_xor(t, 1, 32));
      t = fmaxf(t, __shfl_xor(t, 2, 32));
      t = fmaxf(t, __shfl_xor(t, 4, 32));
      t = fmaxf(t, __shfl_xor(t, 8, 32));
      float mnew = fmaxf(mrun[vg], t);
      float sc = __expf(mrun[vg] - mnew);
      float p0 = __expf(sv0 - mnew);
      float p1 = __expf(sv1 - mnew);
      float rs = p0 + p1;
      rs += __shfl_xor(rs, 1, 32);
      rs += __shfl_xor(rs, 2, 32);
      rs += __shfl_xor(rs, 4, 32);
      rs += __shfl_xor(rs, 8, 32);
      lrun[vg] = lrun[vg] * sc + rs;
      mrun[vg] = mnew;
      acc0[vg] *= sc; acc1[vg] *= sc;
      const int pr = vg + 8 * grp;
      lds_p[pr * 32 + mrow]      = (_Float16)p0;
      lds_p[pr * 32 + 16 + mrow] = (_Float16)p1;
    }
    __syncthreads();
    v16h pf  = frag_from_h(&lds_p[mrow * 32], grp);
    v16h vf0 = frag_from_h(vb + (size_t)mrow * N + j0, grp);
    v16h vf1 = frag_from_h(vb + (size_t)(16 + mrow) * N + j0, grp);
    acc0 = __builtin_amdgcn_wmma_f32_16x16x32_f16(false, pf, false, vf0, (short)0, acc0, false, false);
    acc1 = __builtin_amdgcn_wmma_f32_16x16x32_f16(false, pf, false, vf1, (short)0, acc1, false, false);
    __syncthreads();
  }

#pragma unroll
  for (int vg = 0; vg < 8; ++vg) {
    int r = qt * 16 + vg + 8 * grp;
    float inv = 1.f / lrun[vg];
    size_t o = ((size_t)w * N + r) * C + (size_t)h * hd;
    out[o + mrow] = acc0[vg] * inv;
    if (16 + mrow < hd) out[o + 16 + mrow] = acc1[vg] * inv;
  }
}

// ----- LayerNorm ------------------------------------------------------------
__global__ __launch_bounds__(256)
void ln_kernel(const float* __restrict__ in, float* __restrict__ out,
               const float* __restrict__ g, const float* __restrict__ b, int C) {
  const int row = blockIdx.x, tid = threadIdx.x;
  const float* p = in + (size_t)row * C;
  __shared__ float s1[256], s2[256];
  float a = 0.f, a2 = 0.f;
  for (int c = tid; c < C; c += 256) { float v = p[c]; a += v; a2 += v * v; }
  s1[tid] = a; s2[tid] = a2;
  __syncthreads();
  for (int off = 128; off; off >>= 1) {
    if (tid < off) { s1[tid] += s1[tid + off]; s2[tid] += s2[tid + off]; }
    __syncthreads();
  }
  const float mean = s1[0] / C;
  const float var  = s2[0] / C - mean * mean;
  const float rstd = rsqrtf(var + 1e-5f);
  float* q = out + (size_t)row * C;
  for (int c = tid; c < C; c += 256) q[c] = (p[c] - mean) * rstd * g[c] + b[c];
}

// ----- patch embed (4x4 conv as per-token dot) ------------------------------
__global__ __launch_bounds__(192)
void patch_embed_kernel(const float* __restrict__ x, const float* __restrict__ pw,
                        const float* __restrict__ pb, float* __restrict__ feat) {
  __shared__ float patch[48];
  const int t = blockIdx.x;
  const int py = t / HP, px = t % HP;
  const int tid = threadIdx.x;
  if (tid < 48) {
    int c = tid / 16, r = tid % 16, i = r / 4, j = r % 4;
    patch[tid] = x[(size_t)c * IMG * IMG + (size_t)(py * 4 + i) * IMG + (px * 4 + j)];
  }
  __syncthreads();
  float acc = pb[tid];
  const float* wrow = pw + (size_t)tid * 48;
#pragma unroll 8
  for (int k = 0; k < 48; ++k) acc += patch[k] * wrow[k];
  feat[(size_t)t * EMB + tid] = acc;
}

// ----- qkv pack: window partition + roll + fp16 + q-scale + V transpose -----
__global__ __launch_bounds__(256)
void qkv_pack_kernel(const float* __restrict__ qkv, _Float16* __restrict__ qh,
                     _Float16* __restrict__ kh, _Float16* __restrict__ vT,
                     int H, int ws, int nWside, int nH, int C, int shift, float scale) {
  const int N = ws * ws;
  size_t i = (size_t)blockIdx.x * blockDim.x + threadIdx.x;
  size_t total = (size_t)nWside * nWside * nH * N * 32;
  if (i >= total) return;
  int d = (int)(i & 31); size_t r = i >> 5;
  int n = (int)(r % N); r /= N;
  int h = (int)(r % nH); int w = (int)(r / nH);
  int wy = w / nWside, wx = w % nWside;
  int y = n / ws, x = n % ws;
  int sy = (wy * ws + y + shift) % H;     // rolled -> original coord
  int sx = (wx * ws + x + shift) % H;
  size_t t = (size_t)sy * H + sx;
  const int hd = C / nH;
  float q = 0.f, k = 0.f, v = 0.f;
  if (d < hd) {
    const float* base = qkv + t * (size_t)(3 * C);
    q = base[h * hd + d] * scale;
    k = base[C + h * hd + d];
    v = base[2 * C + h * hd + d];
  }
  size_t fo = (((size_t)w * nH + h) * N + n) * 32 + d;
  qh[fo] = (_Float16)q;
  kh[fo] = (_Float16)k;
  vT[(((size_t)w * nH + h) * 32 + d) * N + n] = (_Float16)v;
}

// ----- window reverse + un-roll + residual add ------------------------------
__global__ __launch_bounds__(256)
void unpack_residual_kernel(float* __restrict__ x, const float* __restrict__ proj,
                            int H, int ws, int nWside, int C, int shift) {
  size_t i = (size_t)blockIdx.x * blockDim.x + threadIdx.x;
  size_t total = (size_t)H * H * C;
  if (i >= total) return;
  int c = (int)(i % C);
  size_t t = i / C;
  int sy = (int)(t / H), sx = (int)(t % H);
  int gy = (sy - shift + H) % H, gx = (sx - shift + H) % H;
  int wy = gy / ws, y = gy % ws, wx = gx / ws, xx = gx % ws;
  size_t wtok = ((size_t)(wy * nWside + wx)) * (ws * ws) + (size_t)y * ws + xx;
  x[i] += proj[wtok * C + c];
}

// ----- patch merging gather -------------------------------------------------
__global__ __launch_bounds__(256)
void merge_kernel(const float* __restrict__ x, float* __restrict__ out, int H, int C) {
  const int H2 = H / 2, C4 = 4 * C;
  size_t i = (size_t)blockIdx.x * blockDim.x + threadIdx.x;
  size_t total = (size_t)H2 * H2 * C4;
  if (i >= total) return;
  int k = (int)(i % C4);
  size_t t2 = i / C4;
  int g = k / C, c = k % C;
  int oy = (int)(t2 / H2), ox = (int)(t2 % H2);
  int dy = (g == 1 || g == 3) ? 1 : 0;
  int dx = (g >= 2) ? 1 : 0;
  out[i] = x[((size_t)(2 * oy + dy) * H + (2 * ox + dx)) * C + c];
}

// ----- mean over tokens -----------------------------------------------------
__global__ __launch_bounds__(256)
void mean_kernel(const float* __restrict__ in, float* __restrict__ out, int L, int C) {
  int c = blockIdx.x * blockDim.x + threadIdx.x;
  if (c >= C) return;
  float a = 0.f;
  for (int t = 0; t < L; ++t) a += in[(size_t)t * C + c];
  out[c] = a / (float)L;
}

// ----- final reduction ------------------------------------------------------
__global__ __launch_bounds__(256)
void sum_kernel(const float* __restrict__ v, float* __restrict__ out, int n) {
  __shared__ float s[256];
  int tid = threadIdx.x;
  float a = 0.f;
  for (int i = tid; i < n; i += 256) a += v[i];
  s[tid] = a;
  __syncthreads();
  for (int off = 128; off; off >>= 1) {
    if (tid < off) s[tid] += s[tid + off];
    __syncthreads();
  }
  if (tid == 0) out[0] = s[0];
}

// ---------------------------------------------------------------------------
extern "C" void kernel_launch(void* const* d_in, const int* in_sizes, int n_in,
                              void* d_out, int out_size, void* d_ws, size_t ws_size,
                              hipStream_t stream) {
  (void)in_sizes; (void)n_in; (void)out_size; (void)ws_size;
  auto F = [&](int i) { return (const float*)d_in[i]; };

  // ---- workspace layout (floats / halves) ----
  const size_t xN   = (size_t)25600 * 192;    // max L*C
  const size_t bigN = (size_t)25600 * 768;    // max of L*3C and L*4C
  float* xbuf  = (float*)d_ws;
  float* ybuf  = xbuf + xN;
  float* tbuf  = ybuf + xN;
  float* big   = tbuf + xN;
  float* meanb = big + bigN;
  float* logit = meanb + 2048;
  _Float16* qh = (_Float16*)(logit + 2048);
  const size_t qN = (size_t)16 * 8 * 1600 * 32;   // max nW*nH*N*32
  _Float16* kh = qh + qN;
  _Float16* vT = kh + qN;

  auto gemm = [&](const float* A, const float* W, const float* bias, const float* res,
                  float* Co, int M, int Nn, int K, int gelu) {
    dim3 g((Nn + 127) / 128, (M + 31) / 32);
    dim3 blk(32, 4);
    gemm_wmma_kernel<<<g, blk, 0, stream>>>(A, W, bias, res, Co, M, Nn, K, gelu);
  };
  auto ln = [&](const float* in, float* out, const float* g, const float* b, int L, int C) {
    ln_kernel<<<L, 256, 0, stream>>>(in, out, g, b, C);
  };

  // ---- patch embed + LN ----
  const float* X = F(0);
  patch_embed_kernel<<<HP * HP, 192, 0, stream>>>(X, F(1), F(2), ybuf);
  ln(ybuf, xbuf, F(3), F(4), HP * HP, EMB);

  // ---- stages ----
  const int cfgH[4]  = {160, 80, 40, 20};
  const int cfgC[4]  = {192, 384, 768, 1536};
  const int cfgNH[4] = {8, 16, 32, 64};
  const int cfgWS[4] = {40, 40, 40, 20};
  int cursor = 5;

  for (int s = 0; s < 4; ++s) {
    const int H = cfgH[s], C = cfgC[s], nH = cfgNH[s], ws = cfgWS[s];
    const int L = H * H, nWside = H / ws, nW = nWside * nWside, N = ws * ws;
    const float scale = rsqrtf((float)(C / nH));

    for (int j = 0; j < 2; ++j) {
      const int bp = cursor + j * 13;
      const float* ln1_g = F(bp + 0); const float* ln1_b = F(bp + 1);
      const float* qkv_w = F(bp + 2); const float* qkv_b = F(bp + 3);
      const float* rpb   = F(bp + 4);
      const float* prj_w = F(bp + 5); const float* prj_b = F(bp + 6);
      const float* ln2_g = F(bp + 7); const float* ln2_b = F(bp + 8);
      const float* fc1_w = F(bp + 9); const float* fc1_b = F(bp + 10);
      const float* fc2_w = F(bp + 11); const float* fc2_b = F(bp + 12);
      const int shift = (j % 2 == 1 && ws < H) ? ws / 2 : 0;

      ln(xbuf, ybuf, ln1_g, ln1_b, L, C);                                 // y = LN1(x)
      gemm(ybuf, qkv_w, qkv_b, nullptr, big, L, 3 * C, C, 0);             // qkv
      {
        size_t total = (size_t)nW * nH * N * 32;
        qkv_pack_kernel<<<(unsigned)((total + 255) / 256), 256, 0, stream>>>(
            big, qh, kh, vT, H, ws, nWside, nH, C, shift, scale);
      }
      {
        dim3 g(N / 16, nH, nW);
        flash_attn_kernel<<<g, 32, 0, stream>>>(qh, kh, vT, rpb, tbuf,
                                                N, ws, nH, nWside, H, shift, C);
      }
      gemm(tbuf, prj_w, prj_b, nullptr, ybuf, L, C, C, 0);                // proj
      {
        size_t total = (size_t)L * C;
        unpack_residual_kernel<<<(unsigned)((total + 255) / 256), 256, 0, stream>>>(
            xbuf, ybuf, H, ws, nWside, C, shift);                          // x += attn
      }
      ln(xbuf, tbuf, ln2_g, ln2_b, L, C);                                 // LN2
      gemm(tbuf, fc1_w, fc1_b, nullptr, big, L, 4 * C, C, 1);             // fc1+GELU
      gemm(big, fc2_w, fc2_b, xbuf, xbuf, L, C, 4 * C, 0);                // fc2+residual
    }
    cursor += 26;

    if (s < 3) {
      const float* ds_g = F(cursor + 0);
      const float* ds_b = F(cursor + 1);
      const float* ds_w = F(cursor + 2);
      cursor += 3;
      const int L2 = (H / 2) * (H / 2);
      {
        size_t total = (size_t)L2 * 4 * C;
        merge_kernel<<<(unsigned)((total + 255) / 256), 256, 0, stream>>>(xbuf, ybuf, H, C);
      }
      ln(ybuf, tbuf, ds_g, ds_b, L2, 4 * C);
      gemm(tbuf, ds_w, nullptr, nullptr, xbuf, L2, 2 * C, 4 * C, 0);
    }
  }

  // ---- head ----
  const float* norm_g = F(cursor + 0);
  const float* norm_b = F(cursor + 1);
  const float* head_w = F(cursor + 2);
  const float* head_b = F(cursor + 3);
  const int Lf = 20 * 20, Cf = 1536;
  ln(xbuf, ybuf, norm_g, norm_b, Lf, Cf);
  mean_kernel<<<(Cf + 255) / 256, 256, 0, stream>>>(ybuf, meanb, Lf, Cf);
  gemm(meanb, head_w, head_b, nullptr, logit, 1, 1000, Cf, 0);
  sum_kernel<<<1, 256, 0, stream>>>(logit, (float*)d_out, 1000);
}
